// AlphaNet_86311662780591
// MI455X (gfx1250) — matrix operand order repaired and checked by reading, not compiled, not run on hardware
//
#include <hip/hip_runtime.h>
#include <hip/hip_bf16.h>

typedef __attribute__((ext_vector_type(16))) _Float16 v16h;
typedef __attribute__((ext_vector_type(8)))  _Float16 v8h;
typedef __attribute__((ext_vector_type(2)))  _Float16 v2h;
typedef __attribute__((ext_vector_type(8)))  float    v8f;
typedef __attribute__((ext_vector_type(4)))  float    v4f;

#define NSAMP  4096
#define FEAT   15
#define WLEN   120
#define SLEN   10
#define NWIN   12
#define NPAIR  105
#define HPROWS 270      // 2*105 + 4*15
#define K1     43200    // 16 * 270 * 10
#define NFC2   512
#define NFC3   128

__device__ __forceinline__ float fillf(float x) {
    // finite -> x, nan/inf -> 0   (x*0 == 0 iff x finite; safe without fast-math)
    return (x * 0.0f == 0.0f) ? x : 0.0f;
}

// ---------------------------------------------------------------------------
// Kernel 1: fp32 -> fp16 weight conversion (fc1_w and fc2_w)
// ---------------------------------------------------------------------------
__global__ void convert_weights_kernel(const float* __restrict__ w1,
                                       const float* __restrict__ w2,
                                       _Float16* __restrict__ w1h,
                                       _Float16* __restrict__ w2h,
                                       int n1, int n2) {
    int i = blockIdx.x * blockDim.x + threadIdx.x;
    int total = n1 + n2;
    for (; i < total; i += gridDim.x * blockDim.x) {
        if (i < n1) w1h[i] = (_Float16)w1[i];
        else        w2h[i - n1] = (_Float16)w2[i - n1];
    }
}

// ---------------------------------------------------------------------------
// Kernel 2: features + BatchNorm + Conv(1,16,(1,3)) + ReLU -> f16 A[4096][43200]
// One workgroup (256 threads) per sample; all intermediates LDS-resident.
// ---------------------------------------------------------------------------
__global__ __launch_bounds__(256)
void feat_conv_kernel(const float* __restrict__ data,
                      const float* __restrict__ bn_gamma, const float* __restrict__ bn_beta,
                      const float* __restrict__ bn_mean,  const float* __restrict__ bn_var,
                      const float* __restrict__ conv_w,   const float* __restrict__ conv_b,
                      _Float16* __restrict__ A) {
    __shared__ float sraw[FEAT * WLEN];     // raw data, overwritten with spread
    __shared__ float sstd[FEAT * NWIN];
    __shared__ float sfeat[HPROWS * NWIN];  // BN-applied feature map
    __shared__ float scw[48];
    __shared__ float scb[16];

    const int tid = threadIdx.x;
    const int n   = blockIdx.x;

    // BN scalars (C == 1): y = x*scale + shift
    const float rs    = rsqrtf(bn_var[0] + 1e-5f);
    const float scale = bn_gamma[0] * rs;
    const float shift = bn_beta[0] - bn_mean[0] * scale;

    if (tid < 48) scw[tid] = conv_w[tid];
    if (tid < 16) scb[tid] = conv_b[tid];
    for (int i = tid; i < FEAT * WLEN; i += 256)
        sraw[i] = data[(size_t)n * FEAT * WLEN + i];
    __syncthreads();

    // per-(feature,window) stats: mean/std/zscore/return/decaylinear
    if (tid < FEAT * NWIN) {
        const int f = tid / NWIN, w = tid % NWIN;
        float* wp = &sraw[f * WLEN + w * SLEN];
        float sum = 0.f;
        #pragma unroll
        for (int s = 0; s < SLEN; ++s) sum += wp[s];
        const float mean = sum * 0.1f;
        const float ret = fillf(wp[SLEN - 1] / wp[0] - 1.0f);
        float dl = 0.f;
        #pragma unroll
        for (int s = 0; s < SLEN; ++s) dl += wp[s] * (float)(s + 1);
        dl *= (1.0f / 55.0f);
        float ss = 0.f;
        #pragma unroll
        for (int s = 0; s < SLEN; ++s) { float sp = wp[s] - mean; wp[s] = sp; ss += sp * sp; }
        const float sd = sqrtf(ss * (1.0f / 9.0f));   // unbiased std
        sstd[tid] = sd;
        const float zs = fillf(mean / sd);
        sfeat[(2 * NPAIR            + f) * NWIN + w] = fillf(sd) * scale + shift;
        sfeat[(2 * NPAIR +     FEAT + f) * NWIN + w] = zs        * scale + shift;
        sfeat[(2 * NPAIR + 2 * FEAT + f) * NWIN + w] = ret       * scale + shift;
        sfeat[(2 * NPAIR + 3 * FEAT + f) * NWIN + w] = fillf(dl) * scale + shift;
    }
    __syncthreads();

    // pairwise cov / corr (105 pairs x 12 windows)
    for (int idx = tid; idx < NPAIR * NWIN; idx += 256) {
        const int pr = idx / NWIN, w = idx % NWIN;
        int i = 0, base = 0, cnt = FEAT - 1;
        while (pr >= base + cnt) { base += cnt; --cnt; ++i; }
        const int j = (pr - base) + i + 1;
        const float* pi = &sraw[i * WLEN + w * SLEN];
        const float* pj = &sraw[j * WLEN + w * SLEN];
        float cv = 0.f;
        #pragma unroll
        for (int s = 0; s < SLEN; ++s) cv += pi[s] * pj[s];
        cv *= (1.0f / 9.0f);
        const float corr = fillf(cv / (sstd[i * NWIN + w] * sstd[j * NWIN + w]) * 0.9f);
        sfeat[pr * NWIN + w]            = corr      * scale + shift;
        sfeat[(NPAIR + pr) * NWIN + w]  = fillf(cv) * scale + shift;
    }
    __syncthreads();

    // Conv(1x3) + ReLU -> f16, two adjacent columns per iteration (b32 stores)
    _Float16* Arow = A + (size_t)n * K1;
    for (int idx = tid; idx < 16 * HPROWS * 5; idx += 256) {
        const int oc  = idx / (HPROWS * 5);
        const int rem = idx % (HPROWS * 5);
        const int r   = rem / 5;
        const int c   = (rem % 5) * 2;
        const float* fr = &sfeat[r * NWIN];
        const float w0 = scw[oc * 3], w1 = scw[oc * 3 + 1], w2 = scw[oc * 3 + 2];
        const float bb = scb[oc];
        const float v0 = fmaxf(bb + w0 * fr[c]     + w1 * fr[c + 1] + w2 * fr[c + 2], 0.f);
        const float v1 = fmaxf(bb + w0 * fr[c + 1] + w1 * fr[c + 2] + w2 * fr[c + 3], 0.f);
        v2h hv; hv.x = (_Float16)v0; hv.y = (_Float16)v1;
        *(v2h*)(Arow + oc * 2700 + r * 10 + c) = hv;
    }
}

// ---------------------------------------------------------------------------
// Kernel 3/4: WMMA GEMM  out = act(A[MxK] * B[NxK]^T + bias)
//   ACT==0: ReLU -> f16 out ;  ACT==1: sigmoid -> f32 out
// WG = 256 threads (8 waves). WG tile 128(M) x 64(N); wave tile 32x32 = 2x2
// of v_wmma_f32_16x16x32_f16, K-step 64 (two 32-K halves -> 8 WMMA/wave/iter).
// Double-buffered LDS staging, pipeline unrolled by 2 so each step uses a
// COMPILE-TIME buffer -> static ds offsets, no per-iteration buffer branches.
// One barrier per K-step; next tile's global loads issued before compute.
// ---------------------------------------------------------------------------
template<int ACT>
__global__ __launch_bounds__(256)
void gemm_wmma_kernel(const _Float16* __restrict__ A, const _Float16* __restrict__ B,
                      const float* __restrict__ bias,
                      _Float16* __restrict__ outH, float* __restrict__ outF,
                      int M, int N, int K) {
    constexpr int KS  = 64;                 // K-step
    constexpr int LDT = KS + 8;             // padded stride: 72 halfwords = 144B
    __shared__ _Float16 ldsA[2][128 * LDT];
    __shared__ _Float16 ldsB[2][64 * LDT];

    const int tid  = threadIdx.x;
    const int lane = tid & 31;
    const int wave = tid >> 5;
    const int wm   = wave & 3;              // 0..3 -> 32-row strip
    const int wn   = wave >> 2;             // 0..1 -> 32-col strip
    const int half = lane >> 4;
    const int l16  = lane & 15;
    const int mBase = blockIdx.x * 128;
    const int nBase = blockIdx.y * 64;

    // staging decomposition: 16B chunks; row = (tid>>3)+32p, kchunk = tid&7
    const int crow = tid >> 3;              // 0..31
    const int kc8  = (tid & 7) * 8;         // halfword offset within row

    const _Float16* pa[4];
    #pragma unroll
    for (int p = 0; p < 4; ++p)
        pa[p] = A + (size_t)(mBase + crow + 32 * p) * K + kc8;
    const _Float16* pb[2];
    #pragma unroll
    for (int p = 0; p < 2; ++p)
        pb[p] = B + (size_t)(nBase + crow + 32 * p) * K + kc8;

    int sA[4], sB[2];
    #pragma unroll
    for (int p = 0; p < 4; ++p) sA[p] = (crow + 32 * p) * LDT + kc8;
    #pragma unroll
    for (int p = 0; p < 2; ++p) sB[p] = (crow + 32 * p) * LDT + kc8;

    int fA[2], fB[2];
    #pragma unroll
    for (int i = 0; i < 2; ++i) fA[i] = (wm * 32 + i * 16 + l16) * LDT;
    #pragma unroll
    for (int j = 0; j < 2; ++j) fB[j] = (wn * 32 + j * 16 + l16) * LDT;

    const v8f vz = {0.f, 0.f, 0.f, 0.f, 0.f, 0.f, 0.f, 0.f};
    v8f acc[2][2] = {{vz, vz}, {vz, vz}};

    // one pipeline step on a compile-time-selected buffer pair
    auto do_step = [&](const _Float16* Ac, const _Float16* Bc,
                       _Float16* An, _Float16* Bn, bool more) {
        v4f ra[4], rb[2];
        if (more) {                          // prefetch next tile into registers
            #pragma unroll
            for (int p = 0; p < 4; ++p) { pa[p] += KS; ra[p] = *(const v4f*)pa[p]; }
            #pragma unroll
            for (int p = 0; p < 2; ++p) { pb[p] += KS; rb[p] = *(const v4f*)pb[p]; }
        }
        // compute on current buffer: two 32-wide K halves
        #pragma unroll
        for (int h = 0; h < 2; ++h) {
            const int ko = 32 * h;
            v16h af[2], bf[2];
            #pragma unroll
            for (int i = 0; i < 2; ++i) {
                // A 16-bit frag: half=0 -> K ko+0-7 & ko+16-23 ; half=1 -> +8
                const _Float16* ap = Ac + fA[i] + ko;
                const v8h lo = *(const v8h*)(ap + 8 * half);
                const v8h hi = *(const v8h*)(ap + 16 + 8 * half);
                #pragma unroll
                for (int e = 0; e < 8; ++e) { af[i][e] = lo[e]; af[i][8 + e] = hi[e]; }
            }
            #pragma unroll
            for (int j = 0; j < 2; ++j) {
                // B 16-bit frag: half=0 -> K ko+0-7 & ko+16-23 ; half=1 -> +8
                const _Float16* bp = Bc + fB[j] + ko;
                const v8h lo = *(const v8h*)(bp + 8 * half);
                const v8h hi = *(const v8h*)(bp + 16 + 8 * half);
                #pragma unroll
                for (int e = 0; e < 8; ++e) { bf[j][e] = lo[e]; bf[j][8 + e] = hi[e]; }
            }
            #pragma unroll
            for (int i = 0; i < 2; ++i)
                #pragma unroll
                for (int j = 0; j < 2; ++j)
                    acc[i][j] = __builtin_amdgcn_wmma_f32_16x16x32_f16(
                        false, af[i], false, bf[j], (short)0, acc[i][j], false, false);
        }
        if (more) {                          // drain prefetched tile -> other buffer
            #pragma unroll
            for (int p = 0; p < 4; ++p) *(v4f*)(An + sA[p]) = ra[p];
            #pragma unroll
            for (int p = 0; p < 2; ++p) *(v4f*)(Bn + sB[p]) = rb[p];
        }
        __syncthreads();                     // single barrier per K-step
    };

    // prologue: stage tile 0 into buffer 0
    #pragma unroll
    for (int p = 0; p < 4; ++p) *(v4f*)&ldsA[0][sA[p]] = *(const v4f*)pa[p];
    #pragma unroll
    for (int p = 0; p < 2; ++p) *(v4f*)&ldsB[0][sB[p]] = *(const v4f*)pb[p];
    __syncthreads();

    const int nk = K / KS;
    int kt = 0;
    for (;;) {                               // unrolled-by-2 ping-pong, static bufs
        do_step(&ldsA[0][0], &ldsB[0][0], &ldsA[1][0], &ldsB[1][0], kt + 1 < nk);
        if (++kt == nk) break;
        do_step(&ldsA[1][0], &ldsB[1][0], &ldsA[0][0], &ldsB[0][0], kt + 1 < nk);
        if (++kt == nk) break;
    }

    // epilogue: C/D layout -> lane l16 = column, vgpr r = row (half*8 + r)
    #pragma unroll
    for (int i = 0; i < 2; ++i) {
        #pragma unroll
        for (int j = 0; j < 2; ++j) {
            const int nn = nBase + wn * 32 + j * 16 + l16;
            const float bv = bias[nn];
            #pragma unroll
            for (int r = 0; r < 8; ++r) {
                const int mm = mBase + wm * 32 + i * 16 + half * 8 + r;
                float v = acc[i][j][r] + bv;
                if (ACT == 0) {
                    v = fmaxf(v, 0.f);
                    outH[(size_t)mm * N + nn] = (_Float16)v;
                } else {
                    v = 1.0f / (1.0f + __expf(-v));
                    outF[(size_t)mm * N + nn] = v;
                }
            }
        }
    }
}

// ---------------------------------------------------------------------------
// Kernel 5: fc3 — out[n] = dot(X2[n,:], w3) + b3
// ---------------------------------------------------------------------------
__global__ __launch_bounds__(256)
void fc3_kernel(const float* __restrict__ X2, const float* __restrict__ w3,
                const float* __restrict__ b3, float* __restrict__ out) {
    __shared__ float sw[NFC3];
    const int tid = threadIdx.x;
    if (tid < NFC3) sw[tid] = w3[tid];
    __syncthreads();
    const int n = blockIdx.x * blockDim.x + tid;
    if (n < NSAMP) {
        const float* row = X2 + (size_t)n * NFC3;
        float s = 0.f;
        #pragma unroll 8
        for (int k = 0; k < NFC3; ++k) s += row[k] * sw[k];
        out[n] = s + b3[0];
    }
}

// ---------------------------------------------------------------------------
extern "C" void kernel_launch(void* const* d_in, const int* in_sizes, int n_in,
                              void* d_out, int out_size, void* d_ws, size_t ws_size,
                              hipStream_t stream) {
    (void)in_sizes; (void)n_in; (void)out_size; (void)ws_size;
    const float* data  = (const float*)d_in[0];
    const float* bng   = (const float*)d_in[1];
    const float* bnb   = (const float*)d_in[2];
    const float* bnm   = (const float*)d_in[3];
    const float* bnv   = (const float*)d_in[4];
    const float* convw = (const float*)d_in[5];
    const float* convb = (const float*)d_in[6];
    const float* fc1w  = (const float*)d_in[7];
    const float* fc1b  = (const float*)d_in[8];
    const float* fc2w  = (const float*)d_in[9];
    const float* fc2b  = (const float*)d_in[10];
    const float* fc3w  = (const float*)d_in[11];
    const float* fc3b  = (const float*)d_in[12];
    float* out = (float*)d_out;

    char* ws = (char*)d_ws;
    const size_t offA  = 0;
    const size_t offW1 = offA  + (size_t)NSAMP * K1 * 2;    // A   f16 [4096][43200]
    const size_t offW2 = offW1 + (size_t)NFC2  * K1 * 2;    // W1h f16 [512][43200]
    const size_t offX1 = offW2 + (size_t)NFC3  * NFC2 * 2;  // W2h f16 [128][512]
    const size_t offX2 = offX1 + (size_t)NSAMP * NFC2 * 2;  // X1  f16 [4096][512]
    _Float16* A   = (_Float16*)(ws + offA);
    _Float16* W1h = (_Float16*)(ws + offW1);
    _Float16* W2h = (_Float16*)(ws + offW2);
    _Float16* X1  = (_Float16*)(ws + offX1);
    float*    X2  = (float*)   (ws + offX2);                // f32 [4096][128]

    convert_weights_kernel<<<4096, 256, 0, stream>>>(fc1w, fc2w, W1h, W2h,
                                                     NFC2 * K1, NFC3 * NFC2);
    feat_conv_kernel<<<NSAMP, 256, 0, stream>>>(data, bng, bnb, bnm, bnv,
                                                convw, convb, A);
    gemm_wmma_kernel<0><<<dim3(NSAMP / 128, NFC2 / 64), 256, 0, stream>>>(
        A, W1h, fc1b, X1, nullptr, NSAMP, NFC2, K1);
    gemm_wmma_kernel<1><<<dim3(NSAMP / 128, NFC3 / 64), 256, 0, stream>>>(
        X1, W2h, fc2b, nullptr, X2, NSAMP, NFC3, NFC2);
    fc3_kernel<<<(NSAMP + 255) / 256, 256, 0, stream>>>(X2, fc3w, fc3b, out);
}
